// RCT_72679436583330
// MI455X (gfx1250) — compile-verified, matrix operand order checked by reading, not run on hardware
//
#include <hip/hip_runtime.h>

// ---------------------------------------------------------------------------
// Types for CDNA5 WMMA (wave32): v_wmma_f32_16x16x32_bf16
// ---------------------------------------------------------------------------
typedef __attribute__((ext_vector_type(16))) __bf16 v16bf;
typedef __attribute__((ext_vector_type(8)))  float  v8f;
typedef __attribute__((ext_vector_type(4)))  float  v4f;

union V16 { v16bf v; v4f f[2]; };

#define DEV __device__ __forceinline__

// Load one 16x32 bf16 operand tile (A or B) in the ISA-mandated VGPR layout:
// lane owns row (lane&15); lanes 0-15 hold K {0..7,16..23}, lanes 16-31 hold
// K {8..15,24..31}  ->  two 16-byte loads per lane.
DEV v16bf load_tile(const __bf16* base, int ld, int row, int k, int lane) {
  const __bf16* p = base + (size_t)row * ld + k + ((lane & 16) ? 8 : 0);
  V16 u;
  u.f[0] = *(const v4f*)(p);
  u.f[1] = *(const v4f*)(p + 16);
  return u.v;
}

DEV v8f bwmma(v16bf a, v16bf b, v8f c) {
  // (neg_a, A, neg_b, B, c_mod, C, reuse_a, reuse_b)
  return __builtin_amdgcn_wmma_f32_16x16x32_bf16(false, a, false, b, (short)0, c,
                                                 false, false);
}

// ---------------------------------------------------------------------------
// f32 -> bf16 conversion
// ---------------------------------------------------------------------------
__global__ __launch_bounds__(256)
void cvt_kernel(const float* __restrict__ in, __bf16* __restrict__ out, int n) {
  int i = blockIdx.x * 256 + threadIdx.x;
  if (i < n) out[i] = (__bf16)in[i];
}

// ---------------------------------------------------------------------------
// GEMM:  Out[M,N] = A[M,K] * W[N,K]^T  (both bf16 row-major, K contiguous)
// optional +Res (f32), optional ReLU, optional f32 and/or bf16 outputs.
// Block = 4 waves; each wave computes a 32x64 tile (2x4 WMMA accum tiles).
// ---------------------------------------------------------------------------
__global__ __launch_bounds__(128)
void gemm_bf16_kernel(const __bf16* __restrict__ A, const __bf16* __restrict__ W,
                      const float* __restrict__ Res, float* __restrict__ OutF,
                      __bf16* __restrict__ OutH, int M, int N, int K, int relu)
{
  const int lane = threadIdx.x & 31;
  const int wave = threadIdx.x >> 5;
  const int m0 = blockIdx.x * 32;
  const int n0 = blockIdx.y * 256 + wave * 64;
  const int rlo = lane & 15;

  v8f zero = {};
  v8f acc[2][4];
#pragma unroll
  for (int i = 0; i < 2; ++i)
#pragma unroll
    for (int j = 0; j < 4; ++j) acc[i][j] = zero;

  for (int k = 0; k < K; k += 32) {
    v16bf a0 = load_tile(A, K, m0 + rlo, k, lane);
    v16bf a1 = load_tile(A, K, m0 + 16 + rlo, k, lane);
#pragma unroll
    for (int j = 0; j < 4; ++j) {
      v16bf b = load_tile(W, K, n0 + j * 16 + rlo, k, lane);
      acc[0][j] = bwmma(a0, b, acc[0][j]);
      acc[1][j] = bwmma(a1, b, acc[1][j]);
    }
  }

  // C/D layout: lane -> (N = lane&15), rows rowOff..rowOff+7 across 8 VGPRs
  const int rowOff = (lane & 16) ? 8 : 0;
#pragma unroll
  for (int i = 0; i < 2; ++i) {
#pragma unroll
    for (int j = 0; j < 4; ++j) {
#pragma unroll
      for (int r = 0; r < 8; ++r) {
        const size_t m = (size_t)(m0 + i * 16 + rowOff + r);
        const size_t n = (size_t)(n0 + j * 16 + rlo);
        float v = acc[i][j][r];
        if (Res)  v += Res[m * N + n];
        if (relu) v = fmaxf(v, 0.f);
        if (OutF) OutF[m * N + n] = v;
        if (OutH) OutH[m * N + n] = (__bf16)v;
      }
    }
  }
}

// ---------------------------------------------------------------------------
// V transpose: v16[(b*L+l)*C + h*128 + dd]  ->  Vt[((b*2+h)*128 + dd)*L + l]
// so the PV B-operand is K-contiguous.
// ---------------------------------------------------------------------------
__global__ __launch_bounds__(256)
void vtrans_kernel(const __bf16* __restrict__ Vin, __bf16* __restrict__ Vt) {
  int idx = blockIdx.x * 256 + threadIdx.x;       // total 2*2*128*4096 = 2M
  int l  = idx & 4095;
  int dd = (idx >> 12) & 127;
  int bh = idx >> 19;
  int b = bh >> 1, h = bh & 1;
  Vt[idx] = Vin[((size_t)b * 4096 + l) * 256 + h * 128 + dd];
}

// ---------------------------------------------------------------------------
// Flash attention, one wave per 16-row Q tile per (b,h).
// S = Q K^T / sqrt(128) with online softmax, O accumulated over 32-key blocks.
// ---------------------------------------------------------------------------
__global__ __launch_bounds__(32)
void flash_kernel(const __bf16* __restrict__ Q, const __bf16* __restrict__ Km,
                  const __bf16* __restrict__ Vt, __bf16* __restrict__ Ctx)
{
  constexpr int L = 4096, C = 256, D = 128;
  __shared__ __attribute__((aligned(16))) __bf16 Pl[16 * 32];

  const int lane = threadIdx.x;
  const int col  = lane & 15;
  const int rowOff = (lane & 16) ? 8 : 0;
  const int bh = blockIdx.y;
  const int b = bh >> 1, h = bh & 1;
  const int q0 = blockIdx.x * 16;

  const __bf16* Qb = Q  + (size_t)b * L * C + h * D;
  const __bf16* Kb = Km + (size_t)b * L * C + h * D;
  const __bf16* Vb = Vt + (size_t)bh * D * L;

  v16bf qa[4];
#pragma unroll
  for (int kc = 0; kc < 4; ++kc) qa[kc] = load_tile(Qb, C, q0 + col, kc * 32, lane);

  v8f zero = {};
  v8f o[8];
#pragma unroll
  for (int t = 0; t < 8; ++t) o[t] = zero;

  float mrow[8], lrow[8];
#pragma unroll
  for (int r = 0; r < 8; ++r) { mrow[r] = -1e30f; lrow[r] = 0.f; }

  const float sc = 0.088388347648318447f;  // 1/sqrt(128)

  for (int key0 = 0; key0 < L; key0 += 32) {
    // ---- S = Q K^T (16x32 scores = two 16x16 accum tiles, 8 WMMAs) ----
    v8f s0 = zero, s1 = zero;
#pragma unroll
    for (int kc = 0; kc < 4; ++kc) {
      v16bf kb0 = load_tile(Kb, C, key0 + col,      kc * 32, lane);
      v16bf kb1 = load_tile(Kb, C, key0 + 16 + col, kc * 32, lane);
      s0 = bwmma(qa[kc], kb0, s0);
      s1 = bwmma(qa[kc], kb1, s1);
    }

    // ---- online softmax: rows live in half-wave lane groups ----
#pragma unroll
    for (int r = 0; r < 8; ++r) {
      float v0 = s0[r] * sc, v1 = s1[r] * sc;
      float t = fmaxf(v0, v1);
#pragma unroll
      for (int m = 8; m >= 1; m >>= 1) t = fmaxf(t, __shfl_xor(t, m, 32));
      const float mn = fmaxf(mrow[r], t);
      const float p0 = __expf(v0 - mn);
      const float p1 = __expf(v1 - mn);
      float su = p0 + p1;
#pragma unroll
      for (int m = 8; m >= 1; m >>= 1) su += __shfl_xor(su, m, 32);
      const float corr = __expf(mrow[r] - mn);
      lrow[r] = lrow[r] * corr + su;
      mrow[r] = mn;
#pragma unroll
      for (int t2 = 0; t2 < 8; ++t2) o[t2][r] *= corr;
      // stage P into LDS (C-layout -> memory) for A-layout reload
      Pl[(rowOff + r) * 32 + col]      = (__bf16)p0;
      Pl[(rowOff + r) * 32 + 16 + col] = (__bf16)p1;
    }
    __syncthreads();

    // ---- O += P V : P is the 16x32 A operand, Vt gives K-contiguous B ----
    v16bf pa = load_tile(Pl, 32, col, 0, lane);
#pragma unroll
    for (int t = 0; t < 8; ++t) {
      v16bf vb = load_tile(Vb, L, t * 16 + col, key0, lane);
      o[t] = bwmma(pa, vb, o[t]);
    }
    __syncthreads();
  }

  // ---- normalize and write ctx (bf16, feeds Wo projection) ----
#pragma unroll
  for (int r = 0; r < 8; ++r) {
    const float inv = 1.f / lrow[r];
    const size_t m = (size_t)(q0 + rowOff + r);
#pragma unroll
    for (int t = 0; t < 8; ++t)
      Ctx[((size_t)b * L + m) * C + h * D + t * 16 + col] = (__bf16)(o[t][r] * inv);
  }
}

// ---------------------------------------------------------------------------
// LayerNorm over C=256, one wave per row; optional post-LN residual add,
// optional f32 and/or bf16 outputs.
// ---------------------------------------------------------------------------
__global__ __launch_bounds__(256)
void ln_kernel(const float* __restrict__ X, const float* __restrict__ G,
               const float* __restrict__ Bt, const float* __restrict__ Res,
               float* __restrict__ OutF, __bf16* __restrict__ OutH)
{
  const int lane = threadIdx.x & 31;
  const int wave = threadIdx.x >> 5;
  const size_t row = (size_t)blockIdx.x * 8 + wave;
  const float* x = X + row * 256;

  float v[8], s = 0.f, ss = 0.f;
#pragma unroll
  for (int t = 0; t < 8; ++t) {
    float f = x[lane + t * 32];
    v[t] = f; s += f; ss += f * f;
  }
#pragma unroll
  for (int m = 16; m >= 1; m >>= 1) {
    s  += __shfl_xor(s,  m, 32);
    ss += __shfl_xor(ss, m, 32);
  }
  const float mu = s * (1.f / 256.f);
  const float rs = rsqrtf(ss * (1.f / 256.f) - mu * mu + 1e-6f);
#pragma unroll
  for (int t = 0; t < 8; ++t) {
    const int c = lane + t * 32;
    float y = (v[t] - mu) * rs * G[c] + Bt[c];
    if (Res)  y += Res[row * 256 + c];
    if (OutF) OutF[row * 256 + c] = y;
    if (OutH) OutH[row * 256 + c] = (__bf16)y;
  }
}

// ---------------------------------------------------------------------------
// Host orchestration
// ---------------------------------------------------------------------------
extern "C" void kernel_launch(void* const* d_in, const int* in_sizes, int n_in,
                              void* d_out, int out_size, void* d_ws, size_t ws_size,
                              hipStream_t stream)
{
  (void)in_sizes; (void)n_in; (void)out_size; (void)ws_size;
  constexpr int B = 2, L = 4096, C = 256, H = 1024;
  constexpr size_t BL = (size_t)B * L;

  const float* Fm   = (const float*)d_in[0];
  const float* Fs   = (const float*)d_in[1];
  const float* Fq   = (const float*)d_in[2];
  const float* Wq   = (const float*)d_in[4];
  const float* Wk   = (const float*)d_in[5];
  const float* Wv   = (const float*)d_in[6];
  const float* Wo   = (const float*)d_in[7];
  const float* LnG  = (const float*)d_in[8];
  const float* LnB  = (const float*)d_in[9];
  const float* W1   = (const float*)d_in[10];
  const float* W2   = (const float*)d_in[11];
  const float* FlnG = (const float*)d_in[12];
  const float* FlnB = (const float*)d_in[13];

  char* ws = (char*)d_ws;
  size_t off = 0;
  auto take = [&](size_t bytes) -> char* {
    char* p = ws + off;
    off = (off + bytes + 255) & ~(size_t)255;
    return p;
  };

  __bf16* wq16  = (__bf16*)take((size_t)3 * C * C * 2);
  __bf16* wk16  = (__bf16*)take((size_t)3 * C * C * 2);
  __bf16* wv16  = (__bf16*)take((size_t)3 * C * C * 2);
  __bf16* wo16  = (__bf16*)take((size_t)3 * C * C * 2);
  __bf16* w1_16 = (__bf16*)take((size_t)3 * H * C * 2);
  __bf16* w2_16 = (__bf16*)take((size_t)3 * C * H * 2);
  __bf16* Fm16  = (__bf16*)take(BL * C * 2);
  __bf16* Fs16  = (__bf16*)take(BL * C * 2);
  __bf16* Fq16  = (__bf16*)take(BL * C * 2);
  __bf16* css16 = (__bf16*)take(BL * C * 2);
  __bf16* cqq16 = (__bf16*)take(BL * C * 2);
  float*  cssF  = (float*) take(BL * C * 4);
  __bf16* q16   = (__bf16*)take(BL * C * 2);
  __bf16* k16   = (__bf16*)take(BL * C * 2);
  __bf16* v16   = (__bf16*)take(BL * C * 2);
  __bf16* ctx16 = (__bf16*)take(BL * C * 2);
  __bf16* ln16b = (__bf16*)take(BL * C * 2);
  float*  attnF = (float*) take(BL * C * 4);     // also reused as yF
  float*  lnF   = (float*) take(BL * C * 4);
  char*   big   = take(BL * H * 2);              // FFN hidden; Vt aliases it
  __bf16* h16   = (__bf16*)big;
  __bf16* Vt    = (__bf16*)big;
  float*  yF    = attnF;                         // attnF dead once lnF is made

  auto cvt = [&](const float* src, __bf16* dst, size_t n) {
    cvt_kernel<<<dim3((unsigned)((n + 255) / 256)), dim3(256), 0, stream>>>(
        src, dst, (int)n);
  };
  auto gemm = [&](const __bf16* A, const __bf16* Wm, const float* Res,
                  float* OutF, __bf16* OutH, int M, int N, int K, int relu) {
    gemm_bf16_kernel<<<dim3((unsigned)(M / 32), (unsigned)(N / 256)),
                       dim3(128), 0, stream>>>(A, Wm, Res, OutF, OutH, M, N, K, relu);
  };
  auto ln = [&](const float* X, const float* G, const float* Bt, const float* Res,
                float* OutF, __bf16* OutH) {
    ln_kernel<<<dim3((unsigned)(BL / 8)), dim3(256), 0, stream>>>(
        X, G, Bt, Res, OutF, OutH);
  };

  // ---- one-time bf16 conversions ----
  cvt(Fm, Fm16, BL * C);
  cvt(Fs, Fs16, BL * C);
  cvt(Fq, Fq16, BL * C);
  cvt(Wq, wq16, (size_t)3 * C * C);
  cvt(Wk, wk16, (size_t)3 * C * C);
  cvt(Wv, wv16, (size_t)3 * C * C);
  cvt(Wo, wo16, (size_t)3 * C * C);
  cvt(W1, w1_16, (size_t)3 * H * C);
  cvt(W2, w2_16, (size_t)3 * C * H);

  auto run_block = [&](int i, const __bf16* Qin, const __bf16* Kin,
                       const __bf16* Vin, const float* Res,
                       float* outF, __bf16* outH) {
    const size_t wOff  = (size_t)i * C * C;
    const size_t w1Off = (size_t)i * H * C;
    // projections
    gemm(Qin, wq16 + wOff, nullptr, nullptr, q16, (int)BL, C, C, 0);
    gemm(Kin, wk16 + wOff, nullptr, nullptr, k16, (int)BL, C, C, 0);
    gemm(Vin, wv16 + wOff, nullptr, nullptr, v16, (int)BL, C, C, 0);
    vtrans_kernel<<<dim3((unsigned)(BL * C / 256)), dim3(256), 0, stream>>>(v16, Vt);
    // attention
    flash_kernel<<<dim3(L / 16, B * 2), dim3(32), 0, stream>>>(q16, k16, Vt, ctx16);
    // output projection + residual, then LN
    gemm(ctx16, wo16 + wOff, Res, attnF, nullptr, (int)BL, C, C, 0);
    ln(attnF, LnG + i * C, LnB + i * C, nullptr, lnF, ln16b);
    // FFN: relu(W1 x) -> W2 -> LN + residual
    gemm(ln16b, w1_16 + w1Off, nullptr, nullptr, h16, (int)BL, H, C, 1);
    gemm(h16, w2_16 + w1Off, nullptr, yF, nullptr, (int)BL, C, H, 0);
    ln(yF, FlnG + i * C, FlnB + i * C, lnF, outF, outH);
  };

  // css = block0(Q=Fs, K=Fs, V=Fm, res=Fm)
  run_block(0, Fs16, Fs16, Fm16, Fm, cssF, css16);
  // cqq = block1(Q=K=V=res=Fq)  (only bf16 copy needed downstream)
  run_block(1, Fq16, Fq16, Fq16, Fq, nullptr, cqq16);
  // csq = block2(Q=cqq, K=Fs, V=css, res=css) -> d_out (f32)
  run_block(2, cqq16, Fs16, css16, cssF, (float*)d_out, nullptr);
}
